// MultiHeadAttention_34437047779750
// MI455X (gfx1250) — matrix operand order repaired and checked by reading, not compiled, not run on hardware
//
#include <hip/hip_runtime.h>
#include <hip/hip_bf16.h>
#include <stdint.h>

// ---------------------------------------------------------------------------
// MI455X (gfx1250) multi-head attention, bf16 WMMA + TDM (tensor_load_to_lds)
// double-buffered LDS staging.
//   0) convert x, Wq, Wk, Wv, Wo fp32 -> bf16
//   1) Q/K projections (bf16 [B*S,D]); V projection stored per-head
//      transposed Vt[b,h,hd,s]  -- WMMA NT GEMM, 128x128 block tile,
//      TDM-staged LDS tiles, double buffered.
//   2) flash attention: block = 4 waves = 4 q-tiles of ONE head; K/V tiles
//      staged once per block via TDM; online softmax fp32; P reshaped
//      through LDS; P@V is NT thanks to Vt layout.
//   3) out = attn@Wo.T + bo (fp32)
// ---------------------------------------------------------------------------

typedef __attribute__((ext_vector_type(16))) __bf16 v16bf;
typedef __attribute__((ext_vector_type(8)))  float  v8f;

#define DEV __device__ __forceinline__

constexpr int cB  = 2;
constexpr int cS  = 2048;
constexpr int cD  = 2048;
constexpr int cH  = 16;
constexpr int cHD = 128;
constexpr int cM  = cB * cS;   // 4096 rows

#ifdef __has_builtin
#if __has_builtin(__builtin_amdgcn_tensor_load_to_lds)
#define HAVE_TDM 1
#endif
#endif
#ifndef HAVE_TDM
#define HAVE_TDM 0
#endif

union FragU { v16bf v; uint4 q[2]; };

DEV uint16_t f32_to_bf16(float f) {
  uint32_t u = __float_as_uint(f);
  u += 0x7FFFu + ((u >> 16) & 1u);      // round to nearest even
  return (uint16_t)(u >> 16);
}

// A-operand fragment (16x32 bf16 MxK): lane m row m; K = hl*16 .. hl*16+15 contiguous.
DEV v16bf load_fragA(const uint16_t* base, int ld, int lane) {
  const int m = lane & 15, hl = lane >> 4;
  const uint16_t* row = base + (size_t)m * ld;
  FragU f;
  f.q[0] = *(const uint4*)(row + hl * 16);
  f.q[1] = *(const uint4*)(row + hl * 16 + 8);
  return f.v;
}

// B-operand fragment (32x16 bf16 KxN) for NT: lane n = source row n,
// K = hl*16 .. hl*16+15 contiguous.
DEV v16bf load_fragB(const uint16_t* base, int ld, int lane) {
  const int n = lane & 15, hl = lane >> 4;
  const uint16_t* row = base + (size_t)n * ld + hl * 16;
  FragU f;
  f.q[0] = *(const uint4*)(row);
  f.q[1] = *(const uint4*)(row + 8);
  return f.v;
}

DEV v8f wmma_bf16(v16bf a, v16bf b, v8f c) {
  return __builtin_amdgcn_wmma_f32_16x16x32_bf16(
      false, a, false, b, (short)0, c, false, false);
}

DEV float redmax16(float v) {
#pragma unroll
  for (int off = 8; off >= 1; off >>= 1) v = fmaxf(v, __shfl_xor(v, off, 16));
  return v;
}
DEV float redsum16(float v) {
#pragma unroll
  for (int off = 8; off >= 1; off >>= 1) v += __shfl_xor(v, off, 16);
  return v;
}

// ---------------------------------------------------------------------------
// TDM: 2-D tile load global->LDS via Tensor DMA descriptor (D#).
// Group-0/Group-1 packed per CDNA5 ISA 8.3/8.4. data_size=1 (2 bytes/elem).
// 6-arg builtin: (u32x4 g0, i32x8 g1, i32x4 g2, i32x4 g3, i32x8 gx, i32 cpol)
// ---------------------------------------------------------------------------
#if HAVE_TDM
typedef __attribute__((ext_vector_type(4))) unsigned int u32x4;
typedef __attribute__((ext_vector_type(8))) int i32x8;
typedef __attribute__((ext_vector_type(4))) int i32x4;

DEV void tdm_load_2d(const void* gptr, unsigned lds_addr,
                     unsigned dim0, unsigned dim1, unsigned stride0,
                     unsigned tile0, unsigned tile1,
                     unsigned pad_int, unsigned pad_amt) {
  const uint64_t ga = (uint64_t)(uintptr_t)gptr;
  u32x4 g0;
  g0[0] = 1u;                                       // count=1 (valid user D#)
  g0[1] = lds_addr;                                 // lds_addr  [63:32]
  g0[2] = (unsigned)ga;                             // global_addr[95:64]
  g0[3] = ((unsigned)(ga >> 32) & 0x01FFFFFFu)      // global_addr[120:96]
          | (2u << 30);                             // type=2 (image)
  i32x8 g1;
  g1[0] = (int)((1u << 16)                          // data_size = 2 bytes
                | (1u << 20)                        // pad_enable
                | (pad_int << 22)                   // pad_interval code
                | (pad_amt << 25));                 // pad_amount code
  g1[1] = (int)((dim0 & 0xFFFFu) << 16);            // tensor_dim0[15:0]
  g1[2] = (int)((dim0 >> 16) | ((dim1 & 0xFFFFu) << 16));
  g1[3] = (int)((dim1 >> 16) | (tile0 << 16));      // tile_dim0
  g1[4] = (int)(tile1 & 0xFFFFu);                   // tile_dim1 (tile_dim2=0)
  g1[5] = (int)stride0;                             // tensor_dim0_stride lo
  g1[6] = 0;
  g1[7] = 0;
  i32x4 z4 = {0, 0, 0, 0};
  i32x8 z8 = {0, 0, 0, 0, 0, 0, 0, 0};
  __builtin_amdgcn_tensor_load_to_lds(g0, g1, z4, z4, z8, 0);
}
#endif

// ---------------------------------------------------------------------------
// Kernel 0: fp32 -> bf16 conversion, 8 elements / thread
// ---------------------------------------------------------------------------
__global__ __launch_bounds__(256) void convert_bf16_kernel(
    const float* __restrict__ in, uint16_t* __restrict__ out, int n8) {
  int i = blockIdx.x * blockDim.x + threadIdx.x;
  if (i >= n8) return;
  const float4* p = (const float4*)in + (size_t)i * 2;
  float4 a = p[0], b = p[1];
  union { uint16_t u[8]; uint4 q; } o;
  o.u[0] = f32_to_bf16(a.x); o.u[1] = f32_to_bf16(a.y);
  o.u[2] = f32_to_bf16(a.z); o.u[3] = f32_to_bf16(a.w);
  o.u[4] = f32_to_bf16(b.x); o.u[5] = f32_to_bf16(b.y);
  o.u[6] = f32_to_bf16(b.z); o.u[7] = f32_to_bf16(b.w);
  ((uint4*)out)[i] = o.q;
}

// ---------------------------------------------------------------------------
// NT GEMM  C[M,N] = A[M,K]*Bw[N,K]^T + bias, TDM-staged double-buffered LDS.
//   MODE 0: bf16 out; MODE 1: bf16 out per-head transposed; MODE 2: fp32 out
// Block = 4 waves; block tile 128x128; wave tile 64x64 (4x4 WMMA).
// ---------------------------------------------------------------------------
template <int MODE>
__global__ __launch_bounds__(128) void gemm_nt_kernel(
    const uint16_t* __restrict__ A, const uint16_t* __restrict__ Bw,
    const float* __restrict__ bias, void* __restrict__ outp,
    int M, int N, int K) {
  constexpr int LP = 40;   // LDS pitch (ushorts): 32 + 4-dword pad (bank-safe)
  __shared__ __align__(16) uint16_t Asm[2][128 * LP];
  __shared__ __align__(16) uint16_t Bsm[2][128 * LP];
  const int lane = threadIdx.x & 31;
  const int wave = threadIdx.x >> 5;
  const int rowB = blockIdx.y * 128;
  const int colB = blockIdx.x * 128;
  const int rw = (wave >> 1) * 64;
  const int cw = (wave & 1) * 64;
  const int NS = K / 32;

  auto stage = [&](int t, int b) {
#if HAVE_TDM
    if (wave == 0) {
      // 128x32 tiles, pad: every 16 dwords (one 64B row) add 4 dwords -> LP=40
      tdm_load_2d(A + (size_t)rowB * K + t * 32,
                  (unsigned)(uintptr_t)&Asm[b][0], K, M, K, 32, 128, 3, 3);
      tdm_load_2d(Bw + (size_t)colB * K + t * 32,
                  (unsigned)(uintptr_t)&Bsm[b][0], K, N, K, 32, 128, 3, 3);
    }
#else
    for (int c = threadIdx.x; c < 512; c += 128) {
      const int r = c >> 2, cc = (c & 3) * 8;
      *(uint4*)&Asm[b][r * LP + cc] =
          *(const uint4*)(A + (size_t)(rowB + r) * K + t * 32 + cc);
      *(uint4*)&Bsm[b][r * LP + cc] =
          *(const uint4*)(Bw + (size_t)(colB + r) * K + t * 32 + cc);
    }
#endif
  };

  stage(0, 0);
  v8f acc[4][4] = {};
  for (int t = 0; t < NS; ++t) {
    const int b = t & 1;
#if HAVE_TDM
    if (wave == 0) __builtin_amdgcn_s_wait_tensorcnt(0);
#endif
    __syncthreads();
    if (t + 1 < NS) stage(t + 1, b ^ 1);
    v16bf af[4], bf[4];
#pragma unroll
    for (int i = 0; i < 4; ++i)
      af[i] = load_fragA(&Asm[b][(rw + 16 * i) * LP], LP, lane);
#pragma unroll
    for (int j = 0; j < 4; ++j)
      bf[j] = load_fragB(&Bsm[b][(cw + 16 * j) * LP], LP, lane);
#pragma unroll
    for (int i = 0; i < 4; ++i)
#pragma unroll
      for (int j = 0; j < 4; ++j)
        acc[i][j] = wmma_bf16(af[i], bf[j], acc[i][j]);
  }

  const int nl = lane & 15, hl = lane >> 4;
#pragma unroll
  for (int i = 0; i < 4; ++i)
#pragma unroll
    for (int j = 0; j < 4; ++j)
#pragma unroll
      for (int r = 0; r < 8; ++r) {
        const int m  = rowB + rw + 16 * i + r + hl * 8;
        const int nn = colB + cw + 16 * j + nl;
        const float c = acc[i][j][r] + bias[nn];
        if (MODE == 0) {
          ((uint16_t*)outp)[(size_t)m * N + nn] = f32_to_bf16(c);
        } else if (MODE == 1) {
          const int bb = m >> 11, s = m & (cS - 1);
          const int h = nn >> 7, hd = nn & (cHD - 1);
          ((uint16_t*)outp)[(size_t)((bb * cH + h) * cHD + hd) * cS + s] =
              f32_to_bf16(c);
        } else {
          ((float*)outp)[(size_t)m * N + nn] = c;
        }
      }
}

// ---------------------------------------------------------------------------
// Flash attention. Block = 4 waves = 4 consecutive 16-query tiles of the SAME
// (b,h). K/V tiles (32 keys) staged once per block via TDM, double buffered.
// Load/compute phases split so the 16 ds_load_b128 of a phase clause together
// instead of serializing wait->wmma->wait.
// ---------------------------------------------------------------------------
__global__ __launch_bounds__(128) void attn_kernel(
    const uint16_t* __restrict__ Qb, const uint16_t* __restrict__ Kb,
    const uint16_t* __restrict__ Vt, uint16_t* __restrict__ Ob) {
  constexpr int KP = 136;  // K-tile pitch: 128 + 8 (bank-conflict-free)
  constexpr int VP = 40;   // V-tile pitch: 32 + 8
  __shared__ __align__(16) uint16_t Ksm[2][32 * KP];   // 32 keys x 128 hd
  __shared__ __align__(16) uint16_t Vsm[2][128 * VP];  // 128 hd x 32 keys
  __shared__ __align__(16) uint16_t Psm[4][16][32];    // per-wave P staging

  const int lane = threadIdx.x & 31;
  const int wave = threadIdx.x >> 5;
  const int tile = blockIdx.x * 4 + wave;
  const int qt = tile & (cS / 16 - 1);
  const int bh = tile / (cS / 16);                     // same for all 4 waves
  const int h  = bh & (cH - 1);
  const int bb = bh / cH;
  const uint16_t* Qh = Qb + (size_t)bb * cS * cD + h * cHD;
  const uint16_t* Kh = Kb + (size_t)bb * cS * cD + h * cHD;
  const uint16_t* Vh = Vt + (size_t)(bb * cH + h) * cHD * cS;
  const int q0 = qt * 16;
  const int nl = lane & 15, hl = lane >> 4;
  const float sc = 0.08838834764831845f;               // 1/sqrt(HD)

  auto stage = [&](int k0, int b) {
#if HAVE_TDM
    if (wave == 0) {
      // K tile: 32 rows x 128 elems, row 256B = 64 dwords, pad 4 -> KP=136
      tdm_load_2d(Kh + (size_t)k0 * cD, (unsigned)(uintptr_t)&Ksm[b][0],
                  cD, cS, cD, 128, 32, 5, 3);
      // V tile: 128 rows x 32 elems, row 64B = 16 dwords, pad 4 -> VP=40
      tdm_load_2d(Vh + k0, (unsigned)(uintptr_t)&Vsm[b][0],
                  cS, cHD, cS, 32, 128, 3, 3);
    }
#else
    for (int c = threadIdx.x; c < 512; c += 128) {
      const int r = c >> 4, cc = (c & 15) * 8;
      *(uint4*)&Ksm[b][r * KP + cc] =
          *(const uint4*)(Kh + (size_t)(k0 + r) * cD + cc);
    }
    for (int c = threadIdx.x; c < 512; c += 128) {
      const int r = c >> 2, cc = (c & 3) * 8;
      *(uint4*)&Vsm[b][r * VP + cc] =
          *(const uint4*)(Vh + (size_t)r * cS + k0 + cc);
    }
#endif
  };

  v16bf qf[4];
#pragma unroll
  for (int kk = 0; kk < 4; ++kk)
    qf[kk] = load_fragA(Qh + (size_t)q0 * cD + kk * 32, cD, lane);

  v8f acc[8] = {};
  float mrow[8], lrow[8];
#pragma unroll
  for (int r = 0; r < 8; ++r) { mrow[r] = -3.0e38f; lrow[r] = 0.0f; }

  stage(0, 0);
  const int NS = cS / 32;
  for (int t = 0; t < NS; ++t) {
    const int b = t & 1;
#if HAVE_TDM
    if (wave == 0) __builtin_amdgcn_s_wait_tensorcnt(0);
#endif
    __syncthreads();
    if (t + 1 < NS) stage((t + 1) * 32, b ^ 1);

    // ---- load ALL K-fragments first (16 independent ds_load_b128) ----
    v16bf kf[8];
#pragma unroll
    for (int kk = 0; kk < 4; ++kk)
      kf[kk] = load_fragB(&Ksm[b][kk * 32], KP, lane);
#pragma unroll
    for (int kk = 0; kk < 4; ++kk)
      kf[4 + kk] = load_fragB(&Ksm[b][16 * KP + kk * 32], KP, lane);

    // ---- scores: two 16x16 tiles, chained over HD (4 x K=32) ----
    v8f s0 = {}, s1 = {};
#pragma unroll
    for (int kk = 0; kk < 4; ++kk) s0 = wmma_bf16(qf[kk], kf[kk], s0);
#pragma unroll
    for (int kk = 0; kk < 4; ++kk) s1 = wmma_bf16(qf[kk], kf[4 + kk], s1);

    // ---- online softmax ----
#pragma unroll
    for (int r = 0; r < 8; ++r) {
      const float mx   = redmax16(fmaxf(s0[r], s1[r]));
      const float mnew = fmaxf(mrow[r], mx);
      const float rs   = __expf(sc * (mrow[r] - mnew));
      mrow[r] = mnew;
      const float p0 = __expf(sc * (s0[r] - mnew));
      const float p1 = __expf(sc * (s1[r] - mnew));
      lrow[r] = lrow[r] * rs + redsum16(p0 + p1);
      const int m = r + hl * 8;
      Psm[wave][m][nl]      = f32_to_bf16(p0);
      Psm[wave][m][nl + 16] = f32_to_bf16(p1);
#pragma unroll
      for (int j = 0; j < 8; ++j) acc[j][r] *= rs;
    }

    // ---- load P + ALL V-fragments, then 8 chained WMMAs ----
    v16bf pf = load_fragA(&Psm[wave][0][0], 32, lane);
    v16bf vf[8];
#pragma unroll
    for (int j = 0; j < 8; ++j) {
      FragU u;
      const uint16_t* vr = &Vsm[b][(j * 16 + nl) * VP + hl * 16];
      u.q[0] = *(const uint4*)(vr);
      u.q[1] = *(const uint4*)(vr + 8);
      vf[j] = u.v;
    }
#pragma unroll
    for (int j = 0; j < 8; ++j) acc[j] = wmma_bf16(pf, vf[j], acc[j]);
  }

  // ---- finalize ----
#pragma unroll
  for (int j = 0; j < 8; ++j)
#pragma unroll
    for (int r = 0; r < 8; ++r) {
      const float o = acc[j][r] / lrow[r];
      const int s = q0 + r + hl * 8;
      const int d = h * cHD + j * 16 + nl;
      Ob[(size_t)(bb * cS + s) * cD + d] = f32_to_bf16(o);
    }
}

// ---------------------------------------------------------------------------
extern "C" void kernel_launch(void* const* d_in, const int* in_sizes, int n_in,
                              void* d_out, int out_size, void* d_ws, size_t ws_size,
                              hipStream_t stream) {
  const float* x  = (const float*)d_in[0];
  const float* Wq = (const float*)d_in[1];
  const float* bq = (const float*)d_in[2];
  const float* Wk = (const float*)d_in[3];
  const float* bk = (const float*)d_in[4];
  const float* Wv = (const float*)d_in[5];
  const float* bv = (const float*)d_in[6];
  const float* Wo = (const float*)d_in[7];
  const float* bo = (const float*)d_in[8];
  float* out = (float*)d_out;

  char* ws = (char*)d_ws;
  size_t off = 0;
  auto walloc = [&](size_t bytes) -> void* {
    void* p = ws + off;
    off += (bytes + 255) & ~(size_t)255;
    return p;
  };
  uint16_t* xbf   = (uint16_t*)walloc((size_t)cM * cD * 2);
  uint16_t* wqbf  = (uint16_t*)walloc((size_t)cD * cD * 2);
  uint16_t* wkbf  = (uint16_t*)walloc((size_t)cD * cD * 2);
  uint16_t* wvbf  = (uint16_t*)walloc((size_t)cD * cD * 2);
  uint16_t* wobf  = (uint16_t*)walloc((size_t)cD * cD * 2);
  uint16_t* qbf   = (uint16_t*)walloc((size_t)cM * cD * 2);
  uint16_t* kbf   = (uint16_t*)walloc((size_t)cM * cD * 2);
  uint16_t* vtbf  = (uint16_t*)walloc((size_t)cM * cD * 2);
  uint16_t* attnb = (uint16_t*)walloc((size_t)cM * cD * 2);

  {
    const int n8x = cM * cD / 8;
    convert_bf16_kernel<<<n8x / 256, 256, 0, stream>>>(x, xbf, n8x);
    const int n8w = cD * cD / 8;
    convert_bf16_kernel<<<n8w / 256, 256, 0, stream>>>(Wq, wqbf, n8w);
    convert_bf16_kernel<<<n8w / 256, 256, 0, stream>>>(Wk, wkbf, n8w);
    convert_bf16_kernel<<<n8w / 256, 256, 0, stream>>>(Wv, wvbf, n8w);
    convert_bf16_kernel<<<n8w / 256, 256, 0, stream>>>(Wo, wobf, n8w);
  }
  {
    dim3 grid(cD / 128, cM / 128);
    gemm_nt_kernel<0><<<grid, 128, 0, stream>>>(xbf, wqbf, bq, qbf,  cM, cD, cD);
    gemm_nt_kernel<0><<<grid, 128, 0, stream>>>(xbf, wkbf, bk, kbf,  cM, cD, cD);
    gemm_nt_kernel<1><<<grid, 128, 0, stream>>>(xbf, wvbf, bv, vtbf, cM, cD, cD);
  }
  {
    const int ntiles = cB * cH * (cS / 16);
    attn_kernel<<<ntiles / 4, 128, 0, stream>>>(qbf, kbf, vtbf, attnb);
  }
  {
    dim3 grid(cD / 128, cM / 128);
    gemm_nt_kernel<2><<<grid, 128, 0, stream>>>(attnb, wobf, bo, out, cM, cD, cD);
  }
}
